// DGNAgent_18245021074049
// MI455X (gfx1250) — matrix operand order repaired and checked by reading, not compiled, not run on hardware
//
#include <hip/hip_runtime.h>
#include <hip/hip_bf16.h>

typedef __attribute__((ext_vector_type(16))) __bf16 v16bf;
typedef __attribute__((ext_vector_type(8)))  __bf16 v8bf;
typedef __attribute__((ext_vector_type(4)))  __bf16 v4bf;
typedef __attribute__((ext_vector_type(8)))  float  v8f;

#define N_NODES 100000
#define N_EDGES 600000
#define HID 128
#define LDS_ROW 520   // 512 + 8 bf16 pad -> 4-dword bank skew per row

__device__ __forceinline__ v8f wmma_bf16(v16bf a, v16bf b, v8f c) {
  // D = A(16x32 bf16) * B(32x16 bf16) + C(16x16 f32)
  return __builtin_amdgcn_wmma_f32_16x16x32_bf16(
      /*neg_a=*/false, a, /*neg_b=*/false, b,
      /*c_mod=*/(short)0, c, /*reuse_a=*/false, /*reuse_b=*/false);
}

// ---------------- weight prep ----------------
__global__ void k_conv_bf16(const float* __restrict__ s, __bf16* __restrict__ d, int n) {
  int i = blockIdx.x * 256 + threadIdx.x;
  if (i < n) d[i] = (__bf16)s[i];
}

// dst[n*256 + k] = k<128 ? wl[n][k] : wr[n][k-128]   (B laid out n-major, contiguous K)
__global__ void k_pack_stack(const float* __restrict__ wl, const float* __restrict__ wr,
                             __bf16* __restrict__ d) {
  int i = blockIdx.x * 256 + threadIdx.x;
  if (i >= 128 * 256) return;
  int n = i >> 8, k = i & 255;
  float v = (k < 128) ? wl[n * 128 + k] : wr[n * 128 + (k - 128)];
  d[n * 256 + k] = (__bf16)v;
}

// ---------------- degree ----------------
__global__ void k_deg(const int* __restrict__ dst, float* __restrict__ deg) {
  int e = blockIdx.x * 256 + threadIdx.x;
  if (e < N_EDGES) atomicAdd(&deg[dst[e]], 1.0f);
}
__global__ void k_invdeg(float* __restrict__ deg) {
  int i = blockIdx.x * 256 + threadIdx.x;
  if (i < N_NODES) deg[i] = 1.0f / fmaxf(deg[i], 1.0f);
}

// ---------------- edge scatter: agg[dst] += x[src] (fp32 atomics, L2-resident) ----------
__global__ void k_scatter(const __bf16* __restrict__ x, const int* __restrict__ src,
                          const int* __restrict__ dst, float* __restrict__ agg) {
  int t = blockIdx.x * 256 + threadIdx.x;
  int e = t >> 5;
  if (e >= N_EDGES) return;
  int lane = t & 31;
  int s = src[e], d = dst[e];
  v4bf xv = *(const v4bf*)(x + (size_t)s * HID + lane * 4);
  float* a = agg + (size_t)d * HID + lane * 4;
  atomicAdd(a + 0, (float)xv[0]);
  atomicAdd(a + 1, (float)xv[1]);
  atomicAdd(a + 2, (float)xv[2]);
  atomicAdd(a + 3, (float)xv[3]);
}

// ---------------- fused MLP: relu(relu(obs@w1.T+b1)@w2.T+b2) -> feat (bf16) ------------
// Block = 64 threads (2 waves) = 32 nodes. Layer1 (K=5) on VALU into LDS,
// layer2 (K=512) via WMMA, 8 n-tiles per wave.
__global__ __launch_bounds__(64) void k_mlp(
    const float* __restrict__ obs, const float* __restrict__ w1, const float* __restrict__ b1,
    const __bf16* __restrict__ w2b, const float* __restrict__ b2, __bf16* __restrict__ feat) {
  __shared__ __bf16 hsh[32 * LDS_ROW];
  const int t = threadIdx.x;
  const int blk = blockIdx.x * 32;
  {
    int nl = t >> 1, half = t & 1;
    int node = blk + nl;
    float o0 = obs[node * 5 + 0], o1 = obs[node * 5 + 1], o2 = obs[node * 5 + 2],
          o3 = obs[node * 5 + 3], o4 = obs[node * 5 + 4];
    int base = half * 256;
    for (int o = 0; o < 256; ++o) {
      int oo = base + o;
      const float* wrow = w1 + oo * 5;
      float v = b1[oo] + o0 * wrow[0] + o1 * wrow[1] + o2 * wrow[2] + o3 * wrow[3] + o4 * wrow[4];
      hsh[nl * LDS_ROW + oo] = (__bf16)fmaxf(v, 0.0f);
    }
  }
  __syncthreads();
  const int wave = t >> 5, lane = t & 31;
  const int l = lane & 15, h = lane >> 4;
  const int tile = blk + wave * 16;
  v8f zero = {0.f, 0.f, 0.f, 0.f, 0.f, 0.f, 0.f, 0.f};
  v8f acc[8];
#pragma unroll
  for (int nt = 0; nt < 8; ++nt) acc[nt] = zero;
  const __bf16* arow = &hsh[(wave * 16 + l) * LDS_ROW];
  for (int kt = 0; kt < 16; ++kt) {
    int k0 = kt * 32;
    v8bf lo = *(const v8bf*)(arow + k0 + 16 * h);
    v8bf hi = *(const v8bf*)(arow + k0 + 16 * h + 8);
    v16bf a = __builtin_shufflevector(lo, hi, 0, 1, 2, 3, 4, 5, 6, 7,
                                      8, 9, 10, 11, 12, 13, 14, 15);
#pragma unroll
    for (int nt = 0; nt < 8; ++nt) {
      v16bf b = *(const v16bf*)(w2b + (size_t)(nt * 16 + l) * 512 + k0 + 16 * h);
      acc[nt] = wmma_bf16(a, b, acc[nt]);
    }
  }
#pragma unroll
  for (int nt = 0; nt < 8; ++nt) {
    float bia = b2[nt * 16 + l];
#pragma unroll
    for (int r = 0; r < 8; ++r) {
      float v = fmaxf(acc[nt][r] + bia, 0.0f);
      feat[(size_t)(tile + r + 8 * h) * HID + nt * 16 + l] = (__bf16)v;
    }
  }
}

// ---------------- SAGE GEMM: relu([mean | x] @ [wl;wr].T + bl) -------------------------
// Wave per 16-node tile. K = 256 (mean 0..127 scaled from fp32 agg, x 128..255 bf16).
__global__ __launch_bounds__(256) void k_sage(
    const float* __restrict__ agg, const float* __restrict__ invdeg,
    const __bf16* __restrict__ x, const __bf16* __restrict__ wst,
    const float* __restrict__ bias, __bf16* __restrict__ out) {
  const int wave = threadIdx.x >> 5;
  const int tile = blockIdx.x * 8 + wave;
  if (tile >= N_NODES / 16) return;
  const int lane = threadIdx.x & 31;
  const int l = lane & 15, h = lane >> 4;
  const int row = tile * 16 + l;
  const float s = invdeg[row];
  v8f zero = {0.f, 0.f, 0.f, 0.f, 0.f, 0.f, 0.f, 0.f};
  v8f acc[8];
#pragma unroll
  for (int nt = 0; nt < 8; ++nt) acc[nt] = zero;
  // K half 1: mean = agg * (1/deg), converted to bf16 in the A-load
#pragma unroll
  for (int kt = 0; kt < 4; ++kt) {
    int k0 = kt * 32;
    const float* ap = agg + (size_t)row * HID + k0 + 16 * h;
    v16bf a;
#pragma unroll
    for (int i = 0; i < 8; ++i) a[i] = (__bf16)(ap[i] * s);
#pragma unroll
    for (int i = 0; i < 8; ++i) a[8 + i] = (__bf16)(ap[8 + i] * s);
#pragma unroll
    for (int nt = 0; nt < 8; ++nt) {
      v16bf b = *(const v16bf*)(wst + (size_t)(nt * 16 + l) * 256 + k0 + 16 * h);
      acc[nt] = wmma_bf16(a, b, acc[nt]);
    }
  }
  // K half 2: root features x (bf16)
#pragma unroll
  for (int kt = 0; kt < 4; ++kt) {
    int kx = kt * 32;
    const __bf16* xp = x + (size_t)row * HID + kx + 16 * h;
    v8bf lo = *(const v8bf*)xp;
    v8bf hi = *(const v8bf*)(xp + 8);
    v16bf a = __builtin_shufflevector(lo, hi, 0, 1, 2, 3, 4, 5, 6, 7,
                                      8, 9, 10, 11, 12, 13, 14, 15);
#pragma unroll
    for (int nt = 0; nt < 8; ++nt) {
      v16bf b = *(const v16bf*)(wst + (size_t)(nt * 16 + l) * 256 + 128 + kx + 16 * h);
      acc[nt] = wmma_bf16(a, b, acc[nt]);
    }
  }
#pragma unroll
  for (int nt = 0; nt < 8; ++nt) {
    float bia = bias[nt * 16 + l];
#pragma unroll
    for (int r = 0; r < 8; ++r) {
      float v = fmaxf(acc[nt][r] + bia, 0.0f);
      out[(size_t)(tile * 16 + r + 8 * h) * HID + nt * 16 + l] = (__bf16)v;
    }
  }
}

// ---------------- Q head: concat(feat, rel1, rel2) @ qw.T + qb (fp32 out) --------------
__global__ void k_qhead(const __bf16* __restrict__ f, const __bf16* __restrict__ r1,
                        const __bf16* __restrict__ r2, const float* __restrict__ qw,
                        const float* __restrict__ qb, float* __restrict__ out) {
  int node = blockIdx.x * 256 + threadIdx.x;
  if (node >= N_NODES) return;
  float acc[9];
#pragma unroll
  for (int j = 0; j < 9; ++j) acc[j] = qb[j];
  const __bf16* parts[3] = {f + (size_t)node * HID, r1 + (size_t)node * HID,
                            r2 + (size_t)node * HID};
  for (int p = 0; p < 3; ++p) {
    const __bf16* xp = parts[p];
    for (int c = 0; c < HID; ++c) {
      float xv = (float)xp[c];
#pragma unroll
      for (int j = 0; j < 9; ++j) acc[j] += xv * qw[j * 384 + p * HID + c];
    }
  }
#pragma unroll
  for (int j = 0; j < 9; ++j) out[(size_t)node * 9 + j] = acc[j];
}

extern "C" void kernel_launch(void* const* d_in, const int* in_sizes, int n_in,
                              void* d_out, int out_size, void* d_ws, size_t ws_size,
                              hipStream_t stream) {
  const float* obs = (const float*)d_in[0];
  const int*   ei  = (const int*)d_in[1];      // (2, E) int32 (jax x64 off)
  const int* srcI = ei;
  const int* dstI = ei + N_EDGES;
  const float* w1 = (const float*)d_in[2];
  const float* b1 = (const float*)d_in[3];
  const float* w2 = (const float*)d_in[4];
  const float* b2 = (const float*)d_in[5];
  const float *wl[4], *bl[4], *wr[4];
  for (int i = 0; i < 4; ++i) {
    wl[i] = (const float*)d_in[6 + 3 * i];
    bl[i] = (const float*)d_in[7 + 3 * i];
    wr[i] = (const float*)d_in[8 + 3 * i];
  }
  const float* qw = (const float*)d_in[18];
  const float* qb = (const float*)d_in[19];
  float* out = (float*)d_out;

  // workspace carve-up (~155 MB)
  char* ws = (char*)d_ws;
  size_t off = 0;
  auto alloc = [&](size_t bytes) -> char* {
    char* p = ws + off;
    off += (bytes + 255) & ~(size_t)255;
    return p;
  };
  float*  agg   = (float*)alloc((size_t)N_NODES * HID * sizeof(float));
  float*  deg   = (float*)alloc((size_t)N_NODES * sizeof(float));
  __bf16* featb = (__bf16*)alloc((size_t)N_NODES * HID * 2);
  __bf16* tmpb  = (__bf16*)alloc((size_t)N_NODES * HID * 2);
  __bf16* r1b   = (__bf16*)alloc((size_t)N_NODES * HID * 2);
  __bf16* r2b   = (__bf16*)alloc((size_t)N_NODES * HID * 2);
  __bf16* w2b   = (__bf16*)alloc((size_t)128 * 512 * 2);
  __bf16* wst[4];
  for (int i = 0; i < 4; ++i) wst[i] = (__bf16*)alloc((size_t)128 * 256 * 2);

  // weight prep (bf16, WMMA-friendly n-major K-contiguous layouts)
  k_conv_bf16<<<(128 * 512 + 255) / 256, 256, 0, stream>>>(w2, w2b, 128 * 512);
  for (int i = 0; i < 4; ++i)
    k_pack_stack<<<128, 256, 0, stream>>>(wl[i], wr[i], wst[i]);

  // degree (identical for every layer)
  hipMemsetAsync(deg, 0, (size_t)N_NODES * sizeof(float), stream);
  k_deg<<<(N_EDGES + 255) / 256, 256, 0, stream>>>(dstI, deg);
  k_invdeg<<<(N_NODES + 255) / 256, 256, 0, stream>>>(deg);

  // feature MLP
  k_mlp<<<N_NODES / 32, 64, 0, stream>>>(obs, w1, b1, w2b, b2, featb);

  // 4 SAGE layers (outputs: tmp -> rel1 -> tmp -> rel2)
  const __bf16* xin[4] = {featb, tmpb, r1b, tmpb};
  __bf16* xout[4] = {tmpb, r1b, tmpb, r2b};
  for (int L = 0; L < 4; ++L) {
    hipMemsetAsync(agg, 0, (size_t)N_NODES * HID * sizeof(float), stream);
    k_scatter<<<(N_EDGES * 32) / 256, 256, 0, stream>>>(xin[L], srcI, dstI, agg);
    k_sage<<<(N_NODES / 16 + 7) / 8, 256, 0, stream>>>(agg, deg, xin[L], wst[L], bl[L], xout[L]);
  }

  // Q head
  k_qhead<<<(N_NODES + 255) / 256, 256, 0, stream>>>(featb, r1b, r2b, qw, qb, out);

  (void)in_sizes; (void)n_in; (void)out_size; (void)ws_size;
}
